// AttentionBlock_84224308675050
// MI455X (gfx1250) — matrix operand order repaired and checked by reading, not compiled, not run on hardware
//
#include <hip/hip_runtime.h>
#include <hip/hip_bf16.h>
#include <math.h>

// ---------------------------------------------------------------------------
// Transformer block for MI455X (gfx1250, wave32, WMMA).
//   B=2, T=2048, D=1024, H=16, DH=64
// Pipeline:
//   1) cvt x, wq,wk,wv, w1, w2  f32 -> f16
//   2) Q = x @ Wq^T, K = x @ Wk^T  (scatter to [B,H,T,DH] f16)
//      Vt = (x @ Wv^T)^T           (scatter to [B,H,DH,T] f16)
//   3) fused causal attention per 16-row Q tile -> h = x + attn_out (f16)
//   4) y1 = gelu(h @ W1^T + b1)  (f16)
//   5) out = x + (y1 @ W2^T + b2) (f32, final output)
// GEMM: 128x128 block tile, 8 waves (4x2), wave tile 32x64,
//       double-buffered LDS fed by GLOBAL_LOAD_ASYNC_TO_LDS_B128 (ASYNCcnt)
//       when the toolchain exposes the builtin; sync staging otherwise.
// ---------------------------------------------------------------------------

typedef _Float16 half4  __attribute__((ext_vector_type(4)));
typedef _Float16 half8  __attribute__((ext_vector_type(8)));
typedef _Float16 half16 __attribute__((ext_vector_type(16)));
typedef float    f32x8  __attribute__((ext_vector_type(8)));
typedef int      v4i    __attribute__((ext_vector_type(4)));

#define B_  2
#define T_  2048
#define D_  1024
#define H_  16
#define DH_ 64

#if __has_builtin(__builtin_amdgcn_global_load_async_to_lds_b128)
#define HAS_ASYNC_LDS 1
#else
#define HAS_ASYNC_LDS 0
#endif

static __device__ __forceinline__ f32x8 wmma_f16(half16 a, half16 b, f32x8 c) {
  // D = A(16x32 f16) * B(32x16 f16) + C(16x16 f32)
  return __builtin_amdgcn_wmma_f32_16x16x32_f16(false, a, false, b,
                                                (short)0, c, false, false);
}

#if HAS_ASYNC_LDS
typedef __attribute__((address_space(1))) v4i* gv4i_p;
typedef __attribute__((address_space(3))) v4i* lv4i_p;
static __device__ __forceinline__ void async_ld_b128(const _Float16* g, _Float16* l) {
  __builtin_amdgcn_global_load_async_to_lds_b128((gv4i_p)(g), (lv4i_p)(l), 0, 0);
}
#endif

static __device__ __forceinline__ void wait_async0() {
#if __has_builtin(__builtin_amdgcn_s_wait_asynccnt)
  __builtin_amdgcn_s_wait_asynccnt(0);
#else
  asm volatile("s_wait_asynccnt 0x0" ::: "memory");
#endif
}

// ---------------------------------------------------------------------------
// f32 -> f16 conversion, 4 elements / thread
// ---------------------------------------------------------------------------
__global__ __launch_bounds__(256)
void cvt_f32_to_f16(const float* __restrict__ src, _Float16* __restrict__ dst, int n) {
  int i = (blockIdx.x * 256 + threadIdx.x) * 4;
  if (i + 3 < n) {
    float4 v = *(const float4*)(src + i);
    half4 h;
    h[0] = (_Float16)v.x; h[1] = (_Float16)v.y;
    h[2] = (_Float16)v.z; h[3] = (_Float16)v.w;
    *(half4*)(dst + i) = h;
  }
}

// ---------------------------------------------------------------------------
// Generic C = A(MxK, row-major f16) * B(NxK, row-major f16)^T
// Block tile 128x128, 8 waves in 4x2; wave tile 32x64 (2 m-tiles x 4 n-tiles).
// K staged through LDS in steps of 32, pitch 40 halves (fragment reads touch
// all 64 banks exactly once: 20*row mod 64 distinct for row=0..15).
// MODE 0: scatter f16 to [B,H,T,DH]       (Q, K projections)
// MODE 1: scatter f16 to [B,H,DH,T]       (V^T projection)
// MODE 2: f16 gelu(acc + bias) -> [M,N]   (FFN layer 1)
// MODE 3: f32 acc + bias + resid -> [M,N] (FFN layer 2 + residual)
// ---------------------------------------------------------------------------
template<int MODE>
__global__ __launch_bounds__(256)
void gemm_nt(const _Float16* __restrict__ A, const _Float16* __restrict__ Bm,
             void* __restrict__ dstv, const float* __restrict__ bias,
             const float* __restrict__ resid, int M, int N, int K) {
  constexpr int BM = 128, BN = 128, BK = 32, PITCH = 40;
  constexpr int ASZ = BM * PITCH;   // halves per A buffer (10240 B)
  constexpr int BSZ = BN * PITCH;   // halves per B buffer (10240 B)
#if HAS_ASYNC_LDS
  __shared__ _Float16 a_lds[2 * ASZ];
  __shared__ _Float16 b_lds[2 * BSZ];
#else
  __shared__ _Float16 a_lds[ASZ];
  __shared__ _Float16 b_lds[BSZ];
#endif

  const int tid  = threadIdx.x;
  const int wid  = tid >> 5;
  const int lane = tid & 31;
  const int mi   = lane & 15;        // row/col within 16-tile
  const int hf   = lane >> 4;        // which half of the wave
  const int wm   = (wid >> 1) * 32;  // wave row offset within block tile
  const int wn   = (wid & 1) * 64;   // wave col offset within block tile
  const int rowBase = blockIdx.y * BM;
  const int colBase = blockIdx.x * BN;

  f32x8 acc[2][4];
#pragma unroll
  for (int mt = 0; mt < 2; ++mt)
#pragma unroll
    for (int nt = 0; nt < 4; ++nt)
#pragma unroll
      for (int i = 0; i < 8; ++i) acc[mt][nt][i] = 0.0f;

  // 256 threads stage a 128x32 tile: each thread two b128 chunks of one row.
  const int srow   = tid >> 1;
  const int schunk = (tid & 1) * 16;

  auto compute = [&](int buf) {
    const _Float16* abase = a_lds + buf * ASZ;
    const _Float16* bbase = b_lds + buf * BSZ;
    half16 af[2];
#pragma unroll
    for (int mt = 0; mt < 2; ++mt) {
      const _Float16* arp = abase + (wm + mt * 16 + mi) * PITCH;
      half8 lo = *(const half8*)(arp + 8 * hf);
      half8 hi = *(const half8*)(arp + 16 + 8 * hf);
#pragma unroll
      for (int i = 0; i < 8; ++i) { af[mt][i] = lo[i]; af[mt][8 + i] = hi[i]; }
    }
#pragma unroll
    for (int nt = 0; nt < 4; ++nt) {
      const _Float16* brp = bbase + (wn + nt * 16 + mi) * PITCH;
      half16 bf;
      half8 b0 = *(const half8*)(brp + 8 * hf);
      half8 b1 = *(const half8*)(brp + 16 + 8 * hf);
#pragma unroll
      for (int i = 0; i < 8; ++i) { bf[i] = b0[i]; bf[8 + i] = b1[i]; }
      acc[0][nt] = wmma_f16(af[0], bf, acc[0][nt]);
      acc[1][nt] = wmma_f16(af[1], bf, acc[1][nt]);
    }
  };

  const int nk = K / BK;
#if HAS_ASYNC_LDS
  // Async double-buffered pipeline: stage tile k+1 while computing tile k.
  auto issue = [&](int k0, int buf) {
    const _Float16* ap = A + (size_t)(rowBase + srow) * K + k0 + schunk;
    _Float16* la = a_lds + buf * ASZ + srow * PITCH + schunk;
    async_ld_b128(ap, la);
    async_ld_b128(ap + 8, la + 8);
    const _Float16* bp = Bm + (size_t)(colBase + srow) * K + k0 + schunk;
    _Float16* lb = b_lds + buf * BSZ + srow * PITCH + schunk;
    async_ld_b128(bp, lb);
    async_ld_b128(bp + 8, lb + 8);
  };
  issue(0, 0);
  wait_async0();
  __syncthreads();
  for (int kk = 0; kk < nk; ++kk) {
    const int cur = kk & 1;
    if (kk + 1 < nk) issue((kk + 1) * BK, cur ^ 1);
    compute(cur);
    wait_async0();
    __syncthreads();
  }
#else
  for (int kk = 0; kk < nk; ++kk) {
    const int k0 = kk * BK;
    __syncthreads();
    const _Float16* ap = A + (size_t)(rowBase + srow) * K + k0 + schunk;
    half8 av0 = *(const half8*)ap;
    half8 av1 = *(const half8*)(ap + 8);
    *(half8*)(a_lds + srow * PITCH + schunk)     = av0;
    *(half8*)(a_lds + srow * PITCH + schunk + 8) = av1;
    const _Float16* bp = Bm + (size_t)(colBase + srow) * K + k0 + schunk;
    half8 bv0 = *(const half8*)bp;
    half8 bv1 = *(const half8*)(bp + 8);
    *(half8*)(b_lds + srow * PITCH + schunk)     = bv0;
    *(half8*)(b_lds + srow * PITCH + schunk + 8) = bv1;
    __syncthreads();
    compute(0);
  }
#endif

  // Epilogue. C layout: VGPR r, lanes 0-15: (M=r, N=lane); lanes 16-31: (M=r+8).
#pragma unroll
  for (int mt = 0; mt < 2; ++mt) {
#pragma unroll
    for (int nt = 0; nt < 4; ++nt) {
      const int n = colBase + wn + nt * 16 + mi;
#pragma unroll
      for (int r = 0; r < 8; ++r) {
        const int m = rowBase + wm + mt * 16 + r + 8 * hf;
        float v = acc[mt][nt][r];
        if (MODE == 0) {
          int b = m >> 11, t = m & (T_ - 1);
          int h = n >> 6,  e = n & (DH_ - 1);
          ((_Float16*)dstv)[((size_t)((b * H_ + h) * T_ + t)) * DH_ + e] = (_Float16)v;
        } else if (MODE == 1) {
          int b = m >> 11, t = m & (T_ - 1);
          int h = n >> 6,  e = n & (DH_ - 1);
          ((_Float16*)dstv)[((size_t)((b * H_ + h) * DH_ + e)) * T_ + t] = (_Float16)v;
        } else if (MODE == 2) {
          float z = v + bias[n];
          float g = 0.5f * z * (1.0f + erff(z * 0.70710678118654752f));
          ((_Float16*)dstv)[(size_t)m * N + n] = (_Float16)g;
        } else {
          float z = v + bias[n] + resid[(size_t)m * N + n];
          ((float*)dstv)[(size_t)m * N + n] = z;
        }
      }
    }
  }
}

// ---------------------------------------------------------------------------
// Fused causal attention. One wave per (b, h, 16-row Q tile).
// Online softmax (flash style), causal early-exit on key loop.
// Writes h = x + attn_out as f16 (FFN input).
// ---------------------------------------------------------------------------
__global__ __launch_bounds__(128)
void attn_fused(const _Float16* __restrict__ qh, const _Float16* __restrict__ kh,
                const _Float16* __restrict__ vth, const float* __restrict__ x,
                _Float16* __restrict__ hbuf) {
  __shared__ _Float16 p_lds[4 * 16 * 40];  // per-wave 16x32 P staging tile
  const int wid  = threadIdx.x >> 5;
  const int lane = threadIdx.x & 31;
  const int mi   = lane & 15;
  const int hf   = lane >> 4;

  const int gw = blockIdx.x * 4 + wid;     // 4096 waves total
  const int qt = gw & (T_ / 16 - 1);
  const int h  = (gw >> 7) & (H_ - 1);
  const int b  = gw >> 11;
  const int q0 = qt * 16;

  const _Float16* qbase  = qh  + ((size_t)(b * H_ + h) * T_ + q0) * DH_;
  const _Float16* kbase  = kh  + ((size_t)(b * H_ + h) * T_) * DH_;
  const _Float16* vtbase = vth + ((size_t)(b * H_ + h) * DH_) * T_;

  // Q fragments for K-dim 0..31 and 32..63, softmax scale (1/8) folded in.
  half16 aq0, aq1;
  {
    const _Float16* qrow = qbase + mi * DH_;
    half8 c0 = *(const half8*)(qrow + 8 * hf);
    half8 c1 = *(const half8*)(qrow + 16 + 8 * hf);
    half8 c2 = *(const half8*)(qrow + 32 + 8 * hf);
    half8 c3 = *(const half8*)(qrow + 48 + 8 * hf);
#pragma unroll
    for (int i = 0; i < 8; ++i) {
      aq0[i] = c0[i]; aq0[8 + i] = c1[i];
      aq1[i] = c2[i]; aq1[8 + i] = c3[i];
    }
    const _Float16 sc = (_Float16)0.125f;
#pragma unroll
    for (int i = 0; i < 16; ++i) { aq0[i] *= sc; aq1[i] *= sc; }
  }

  f32x8 accO[4];
  float mx[8], sm[8];
#pragma unroll
  for (int nt = 0; nt < 4; ++nt)
#pragma unroll
    for (int i = 0; i < 8; ++i) accO[nt][i] = 0.0f;
#pragma unroll
  for (int r = 0; r < 8; ++r) { mx[r] = -3.0e38f; sm[r] = 0.0f; }

  _Float16* pw = p_lds + wid * 16 * 40;
  const int kend = q0 + 16;                 // causal: keys <= q0+15
  for (int j0 = 0; j0 < kend; j0 += 32) {
    f32x8 s0, s1;
#pragma unroll
    for (int sub = 0; sub < 2; ++sub) {
      const int kb = j0 + sub * 16;
      const _Float16* krow = kbase + (size_t)(kb + mi) * DH_;
      half16 bk0, bk1;
      {
        half8 e0 = *(const half8*)(krow + 8 * hf);
        half8 e1 = *(const half8*)(krow + 16 + 8 * hf);
        half8 e2 = *(const half8*)(krow + 32 + 8 * hf);
        half8 e3 = *(const half8*)(krow + 48 + 8 * hf);
#pragma unroll
        for (int i = 0; i < 8; ++i) {
          bk0[i] = e0[i]; bk0[8 + i] = e1[i];
          bk1[i] = e2[i]; bk1[8 + i] = e3[i];
        }
      }
      f32x8 s;
#pragma unroll
      for (int i = 0; i < 8; ++i) s[i] = 0.0f;
      s = wmma_f16(aq0, bk0, s);
      s = wmma_f16(aq1, bk1, s);
      // causal mask: key index kb+mi vs query index q0+r+8*hf
#pragma unroll
      for (int r = 0; r < 8; ++r) {
        if (kb + mi > q0 + r + 8 * hf) s[r] = -3.0e38f;
      }
      if (sub == 0) s0 = s; else s1 = s;
    }

    // Online softmax: per-row reduction over the 16-lane half-wave.
#pragma unroll
    for (int r = 0; r < 8; ++r) {
      float rm = fmaxf(s0[r], s1[r]);
#pragma unroll
      for (int o = 8; o >= 1; o >>= 1) rm = fmaxf(rm, __shfl_xor(rm, o, 32));
      float mnew = fmaxf(mx[r], rm);
      float fac  = __expf(mx[r] - mnew);
      float p0   = __expf(s0[r] - mnew);
      float p1   = __expf(s1[r] - mnew);
      float ps   = p0 + p1;
#pragma unroll
      for (int o = 8; o >= 1; o >>= 1) ps += __shfl_xor(ps, o, 32);
      sm[r] = sm[r] * fac + ps;
      mx[r] = mnew;
      s0[r] = p0; s1[r] = p1;
#pragma unroll
      for (int nt = 0; nt < 4; ++nt) accO[nt][r] *= fac;
    }

    // Transpose P (C-layout) -> A-fragment layout via per-wave LDS tile.
#pragma unroll
    for (int r = 0; r < 8; ++r) {
      pw[(r + 8 * hf) * 40 + mi]      = (_Float16)s0[r];
      pw[(r + 8 * hf) * 40 + 16 + mi] = (_Float16)s1[r];
    }
    asm volatile("s_wait_dscnt 0" ::: "memory");
    half16 pa;
    {
      const _Float16* pr = pw + mi * 40;
      half8 l0 = *(const half8*)(pr + 8 * hf);
      half8 l1 = *(const half8*)(pr + 16 + 8 * hf);
#pragma unroll
      for (int i = 0; i < 8; ++i) { pa[i] = l0[i]; pa[8 + i] = l1[i]; }
    }

    // accO += P(16x32) * V(32x64); Vt rows are contiguous along t.
#pragma unroll
    for (int nt = 0; nt < 4; ++nt) {
      const _Float16* vrow = vtbase + (size_t)(nt * 16 + mi) * T_ + j0;
      half16 bv;
      half8 v0 = *(const half8*)(vrow + 8 * hf);
      half8 v1 = *(const half8*)(vrow + 16 + 8 * hf);
#pragma unroll
      for (int i = 0; i < 8; ++i) { bv[i] = v0[i]; bv[8 + i] = v1[i]; }
      accO[nt] = wmma_f16(pa, bv, accO[nt]);
    }
  }

  // Epilogue: normalize, add residual x, store h as f16.
#pragma unroll
  for (int nt = 0; nt < 4; ++nt) {
    const int e = nt * 16 + mi;
#pragma unroll
    for (int r = 0; r < 8; ++r) {
      const int t = q0 + r + 8 * hf;
      float o = accO[nt][r] / sm[r];
      size_t idx = ((size_t)(b * T_ + t)) * D_ + h * DH_ + e;
      hbuf[idx] = (_Float16)(x[idx] + o);
    }
  }
}

// ---------------------------------------------------------------------------
extern "C" void kernel_launch(void* const* d_in, const int* in_sizes, int n_in,
                              void* d_out, int out_size, void* d_ws, size_t ws_size,
                              hipStream_t stream) {
  (void)in_sizes; (void)n_in; (void)out_size; (void)ws_size;
  const float* x  = (const float*)d_in[0];
  // d_in[1] = mask (causal tril) -- implemented analytically
  const float* wq = (const float*)d_in[2];
  const float* wk = (const float*)d_in[3];
  const float* wv = (const float*)d_in[4];
  const float* w1 = (const float*)d_in[5];
  const float* b1 = (const float*)d_in[6];
  const float* w2 = (const float*)d_in[7];
  const float* b2 = (const float*)d_in[8];

  char* ws = (char*)d_ws;
  size_t off = 0;
  auto alloc = [&](size_t bytes) -> void* {
    void* p = ws + off;
    off += (bytes + 255) & ~(size_t)255;
    return p;
  };
  _Float16* xh  = (_Float16*)alloc((size_t)B_ * T_ * D_ * 2);       // 8 MB
  _Float16* wqh = (_Float16*)alloc((size_t)H_ * DH_ * D_ * 2);      // 2 MB
  _Float16* wkh = (_Float16*)alloc((size_t)H_ * DH_ * D_ * 2);
  _Float16* wvh = (_Float16*)alloc((size_t)H_ * DH_ * D_ * 2);
  _Float16* w1h = (_Float16*)alloc((size_t)4 * D_ * D_ * 2);        // 8 MB
  _Float16* w2h = (_Float16*)alloc((size_t)4 * D_ * D_ * 2);        // 8 MB
  _Float16* qh  = (_Float16*)alloc((size_t)B_ * H_ * T_ * DH_ * 2); // 8 MB
  _Float16* kh  = (_Float16*)alloc((size_t)B_ * H_ * T_ * DH_ * 2);
  _Float16* vth = (_Float16*)alloc((size_t)B_ * H_ * DH_ * T_ * 2);
  _Float16* hh  = (_Float16*)alloc((size_t)B_ * T_ * D_ * 2);       // 8 MB
  _Float16* y1h = (_Float16*)alloc((size_t)B_ * T_ * 4 * D_ * 2);   // 32 MB

  auto cvt = [&](const float* s, _Float16* d, int n) {
    int blocks = (n / 4 + 255) / 256;
    cvt_f32_to_f16<<<blocks, 256, 0, stream>>>(s, d, n);
  };
  cvt(x,  xh,  B_ * T_ * D_);
  cvt(wq, wqh, H_ * DH_ * D_);
  cvt(wk, wkh, H_ * DH_ * D_);
  cvt(wv, wvh, H_ * DH_ * D_);
  cvt(w1, w1h, 4 * D_ * D_);
  cvt(w2, w2h, 4 * D_ * D_);

  const int M = B_ * T_;  // 4096
  dim3 blk(256);
  // QKV projections: M=4096, N=1024, K=1024
  gemm_nt<0><<<dim3(D_ / 128, M / 128), blk, 0, stream>>>(xh, wqh, qh, nullptr, nullptr, M, D_, D_);
  gemm_nt<0><<<dim3(D_ / 128, M / 128), blk, 0, stream>>>(xh, wkh, kh, nullptr, nullptr, M, D_, D_);
  gemm_nt<1><<<dim3(D_ / 128, M / 128), blk, 0, stream>>>(xh, wvh, vth, nullptr, nullptr, M, D_, D_);
  // Fused attention: 4096 waves, 4 waves/block
  attn_fused<<<(B_ * H_ * (T_ / 16)) / 4, 128, 0, stream>>>(qh, kh, vth, x, hh);
  // FFN1: M=4096, N=4096, K=1024, gelu
  gemm_nt<2><<<dim3(4 * D_ / 128, M / 128), blk, 0, stream>>>(hh, w1h, y1h, b1, nullptr, M, 4 * D_, D_);
  // FFN2: M=4096, N=1024, K=4096, bias + residual, f32 out
  gemm_nt<3><<<dim3(D_ / 128, M / 128), blk, 0, stream>>>(y1h, w2h, d_out, b2, x, M, D_, 4 * D_);
}
